// LSTMLayer_32160715112737
// MI455X (gfx1250) — matrix operand order repaired and checked by reading, not compiled, not run on hardware
//
#include <hip/hip_runtime.h>
#include <math.h>

#define B_   64
#define T_   512
#define D_   256
#define H_   512
#define KDIM 768
#define GBLK 16        // persistent blocks; block owns H_/GBLK = 32 h-columns
#define HB   32        // h columns per block
#define NC   128       // gate columns per block (4 gates x 32)
#define ZSTR 776       // padded K-stride of z in LDS (multiple of 8 bf16 = 16B)
#define GSTR 132       // padded column count of gate buffer

typedef __attribute__((ext_vector_type(16))) __bf16        v16bf;
typedef __attribute__((ext_vector_type(8)))  float         v8f;
typedef __attribute__((ext_vector_type(8)))  unsigned short v8us;

struct Two8 { v8us lo, hi; };   // 32 bytes == v16bf

__device__ __forceinline__ unsigned short f2bf(float f) {
  unsigned u = __float_as_uint(f);
  u += 0x7FFFu + ((u >> 16) & 1u);          // round-to-nearest-even
  return (unsigned short)(u >> 16);
}

// ---------------------------------------------------------------------------
// Pre-pack the 4 gate weight matrices into bf16 WMMA B-fragment order:
// wfrag index = ((((blk*8 + wave)*24 + kk)*32 + lane)*16 + e)
// where element (lane, e) of K-step kk holds W_g[(kk*32 + (lane>>4)*16 + e)*H + hcol]
// with c = wave*16 + (lane&15), g = c>>5, hcol = blk*32 + (c&31).
// This makes the per-K-step B load one contiguous 32B/lane global load.
// ---------------------------------------------------------------------------
__global__ void pack_w_kernel(const float* __restrict__ Wf, const float* __restrict__ Wi,
                              const float* __restrict__ Wc, const float* __restrict__ Wo,
                              unsigned short* __restrict__ wfrag) {
  int idx = blockIdx.x * 256 + threadIdx.x;      // 0 .. 1572863
  int e    = idx & 15;
  int L    = (idx >> 4) & 31;
  int kk   = (idx >> 9) % 24;
  int rest = (idx >> 9) / 24;                    // blk*8 + wave
  int w    = rest & 7;
  int blk  = rest >> 3;
  int c    = w * 16 + (L & 15);
  int g    = c >> 5;
  int cg   = c & 31;
  int hcol = blk * HB + cg;
  int K    = kk * 32 + (L >> 4) * 16 + e;
  const float* Wg = (g == 0) ? Wf : (g == 1) ? Wi : (g == 2) ? Wc : Wo;
  wfrag[idx] = f2bf(Wg[K * H_ + hcol]);
}

__global__ void init_state_kernel(unsigned short* __restrict__ hbuf, unsigned* __restrict__ bar) {
  int i = blockIdx.x * 256 + threadIdx.x;
  if (i < 2 * B_ * H_) hbuf[i] = 0;              // both h double-buffers -> h0 = 0
  if (i < 2) bar[i] = 0;                         // grid barrier {count, generation}
}

// ---------------------------------------------------------------------------
// Persistent cooperative LSTM kernel: 16 blocks x 256 threads (8 wave32s).
// ---------------------------------------------------------------------------
__global__ __launch_bounds__(256) void lstm_persist_kernel(
    const float*         __restrict__ x,      // [B,T,D] fp32
    const unsigned short* __restrict__ wfrag, // packed bf16 B-fragments
    const float* __restrict__ bfp, const float* __restrict__ bip,
    const float* __restrict__ bcp, const float* __restrict__ bop,
    unsigned short* __restrict__ hbuf,        // 2 x [B,H] bf16 (double buffer)
    float*         __restrict__ out,          // [B,T,H] fp32
    unsigned*      __restrict__ bar) {
  __shared__ __align__(16) unsigned short zlds[B_ * ZSTR];  // z = [x_t | h] bf16
  __shared__ float gbuf[B_ * GSTR];                         // activated gates
  __shared__ float cst[B_ * HB];                            // cell state (whole seq)

  const int tid  = threadIdx.x;
  const int wv   = tid >> 5;
  const int lane = tid & 31;
  const int lm   = lane & 15;
  const int hi2  = lane >> 4;
  const int blk  = blockIdx.x;

  for (int i = tid; i < B_ * HB; i += 256) cst[i] = 0.0f;

  // Per-thread constants for the bias/activation scatter: this thread's gate column.
  const int   gc      = wv * 16 + lm;        // 0..127
  const int   g       = gc >> 5;             // 0=f 1=i 2=c~ 3=o
  const int   cg      = gc & 31;
  const int   hcol    = blk * HB + cg;
  const float bias    = (g == 0 ? bfp : g == 1 ? bip : g == 2 ? bcp : bop)[hcol];
  const bool  useTanh = (g == 2);

  const v16bf* wfv   = (const v16bf*)wfrag;
  const int    wbase = (blk * 8 + wv) * (24 * 32) + lane;

#pragma unroll 1
  for (int t = 0; t < T_; ++t) {
    const unsigned short* hread  = hbuf + (t & 1)       * (B_ * H_);
    unsigned short*       hwrite = hbuf + ((t + 1) & 1) * (B_ * H_);

    // ---- stage z = [x_t | h_{t}] into LDS as bf16 -------------------------
#pragma unroll
    for (int q = 0; q < 8; ++q) {            // x part: 64 rows x 32 chunks(8 f32)
      int ch = q * 256 + tid;
      int row = ch >> 5, cc = ch & 31;
      const float4* ps = (const float4*)(x + ((size_t)row * T_ + t) * D_ + cc * 8);
      float4 a = ps[0], b = ps[1];
      v8us u;
      u[0] = f2bf(a.x); u[1] = f2bf(a.y); u[2] = f2bf(a.z); u[3] = f2bf(a.w);
      u[4] = f2bf(b.x); u[5] = f2bf(b.y); u[6] = f2bf(b.z); u[7] = f2bf(b.w);
      *(v8us*)&zlds[row * ZSTR + cc * 8] = u;
    }
#pragma unroll
    for (int q = 0; q < 16; ++q) {           // h part: 64 rows x 64 chunks(8 bf16)
      int ch = q * 256 + tid;
      int row = ch >> 6, cc = ch & 63;
      uint4 v = *(const uint4*)(hread + row * H_ + cc * 8);
      *(uint4*)&zlds[row * ZSTR + D_ + cc * 8] = v;
    }
    __syncthreads();

    // ---- matmul: wave owns 16 gate columns x all 4 M-tiles ---------------
    v8f acc[4] = {};
#pragma unroll 4
    for (int kk = 0; kk < 24; ++kk) {
      v16bf bfr = wfv[wbase + kk * 32];      // coalesced 32B/lane from L2
      int k0 = kk * 32 + hi2 * 16;
#pragma unroll
      for (int m = 0; m < 4; ++m) {
        const v8us* p = (const v8us*)&zlds[(m * 16 + lm) * ZSTR + k0];
        Two8 t8; t8.lo = p[0]; t8.hi = p[1]; // K chunks [k0,k0+8) and [k0+8,k0+16)
        v16bf av = __builtin_bit_cast(v16bf, t8);
        acc[m] = __builtin_amdgcn_wmma_f32_16x16x32_bf16(
            false, av, false, bfr, (short)0, acc[m], false, false);
      }
    }

    // ---- bias + activation, scatter into gate buffer ---------------------
#pragma unroll
    for (int m = 0; m < 4; ++m) {
#pragma unroll
      for (int r = 0; r < 8; ++r) {
        int row = m * 16 + hi2 * 8 + r;      // C-matrix VGPR layout (ISA 7.12.2)
        float v = acc[m][r] + bias;
        float a = useTanh ? tanhf(v) : 1.0f / (1.0f + __expf(-v));
        gbuf[row * GSTR + gc] = a;
      }
    }
    __syncthreads();

    // ---- elementwise cell update for this block's 32 h-columns -----------
#pragma unroll
    for (int q = 0; q < 8; ++q) {
      int e = q * 256 + tid;                 // 64 rows x 32 cols
      int row = e >> 5, col = e & 31;
      float fg = gbuf[row * GSTR + col];
      float ig = gbuf[row * GSTR + 32 + col];
      float cc = gbuf[row * GSTR + 64 + col];
      float og = gbuf[row * GSTR + 96 + col];
      float cp = cst[row * HB + col];
      float cn = cp * fg + cc * ig;
      float hn = og * tanhf(cn);
      cst[row * HB + col] = cn;
      int oc = blk * HB + col;
      hwrite[row * H_ + oc] = f2bf(hn);
      out[((size_t)row * T_ + t) * H_ + oc] = hn;
    }

    // ---- device-wide barrier: publish h_{t+1}, sync all 16 blocks --------
    __syncthreads();
    if (tid == 0) {
      unsigned gen  = __hip_atomic_load(&bar[1], __ATOMIC_RELAXED, __HIP_MEMORY_SCOPE_AGENT);
      unsigned prev = __hip_atomic_fetch_add(&bar[0], 1u, __ATOMIC_ACQ_REL, __HIP_MEMORY_SCOPE_AGENT);
      if (prev == GBLK - 1) {
        __hip_atomic_store(&bar[0], 0u, __ATOMIC_RELAXED, __HIP_MEMORY_SCOPE_AGENT);
        __hip_atomic_fetch_add(&bar[1], 1u, __ATOMIC_RELEASE, __HIP_MEMORY_SCOPE_AGENT);
      } else {
        while (__hip_atomic_load(&bar[1], __ATOMIC_ACQUIRE, __HIP_MEMORY_SCOPE_AGENT) == gen) {
          __builtin_amdgcn_s_sleep(2);
        }
      }
    }
    __syncthreads();
  }
}

// ---------------------------------------------------------------------------
extern "C" void kernel_launch(void* const* d_in, const int* in_sizes, int n_in,
                              void* d_out, int out_size, void* d_ws, size_t ws_size,
                              hipStream_t stream) {
  const float* x  = (const float*)d_in[0];
  const float* Wf = (const float*)d_in[1];
  const float* bf = (const float*)d_in[2];
  const float* Wi = (const float*)d_in[3];
  const float* bi = (const float*)d_in[4];
  const float* Wo = (const float*)d_in[5];
  const float* bo = (const float*)d_in[6];
  const float* Wc = (const float*)d_in[7];
  const float* bc = (const float*)d_in[8];
  float* out = (float*)d_out;

  char* ws = (char*)d_ws;
  unsigned short* wfrag = (unsigned short*)(ws);                     // 3,145,728 B
  unsigned short* hbuf  = (unsigned short*)(ws + (size_t)3145728);   //   131,072 B
  unsigned*       bar   = (unsigned*)(ws + (size_t)3276800);         //         8 B

  pack_w_kernel<<<1572864 / 256, 256, 0, stream>>>(Wf, Wi, Wc, Wo, wfrag);
  init_state_kernel<<<(2 * B_ * H_) / 256, 256, 0, stream>>>(hbuf, bar);
  lstm_persist_kernel<<<GBLK, 256, 0, stream>>>(x, wfrag, bf, bi, bc, bo,
                                                hbuf, out, bar);
}
